// CacheAttention_21895743275056
// MI455X (gfx1250) — matrix operand order repaired and checked by reading, not compiled, not run on hardware
//
#include <hip/hip_runtime.h>

// ---------------------------------------------------------------------------
// MHA forward for MI455X (gfx1250, wave32, WMMA 16x16x32 bf16)
// T=2048, B=4, E=1024, H=16, D=64
// ---------------------------------------------------------------------------

typedef __attribute__((ext_vector_type(16))) __bf16 v16bf;
typedef __attribute__((ext_vector_type(8)))  __bf16 v8bf;
typedef __attribute__((ext_vector_type(8)))  float  v8f;

#define T_LEN 2048
#define BATCH 4
#define EMB   1024
#define HEADS 16
#define HDIM  64
#define NHEAD (BATCH * HEADS)   // 64
#define MROWS (T_LEN * BATCH)   // 8192

static __device__ __forceinline__ unsigned short f2bf(float f) {
  unsigned u = __float_as_uint(f);
  u += 0x7FFFu + ((u >> 16) & 1u);            // round-to-nearest-even
  return (unsigned short)(u >> 16);
}

static __device__ __forceinline__ v16bf cat8(v8bf lo, v8bf hi) {
  v16bf r;
#pragma unroll
  for (int i = 0; i < 8; ++i) { r[i] = lo[i]; r[i + 8] = hi[i]; }
  return r;
}
static __device__ __forceinline__ v16bf load16(const unsigned short* p) {
  return *reinterpret_cast<const v16bf*>(p);
}
static __device__ __forceinline__ v8bf load8(const unsigned short* p) {
  return *reinterpret_cast<const v8bf*>(p);
}
static __device__ __forceinline__ v8f wmma_bf16(v16bf a, v16bf b, v8f c) {
  return __builtin_amdgcn_wmma_f32_16x16x32_bf16(false, a, false, b,
                                                 (short)0, c, false, false);
}

// ---------------------------------------------------------------------------
// f32 -> bf16 bulk convert (operands converted exactly once; keeps the GEMM
// inner loops free of cvt VALU traffic)
// ---------------------------------------------------------------------------
__global__ void __launch_bounds__(256)
cvt_bf16_kernel(const float* __restrict__ s, unsigned short* __restrict__ d, int n) {
  int i = blockIdx.x * 256 + threadIdx.x;
  if (i < n) d[i] = f2bf(s[i]);
}

// ---------------------------------------------------------------------------
// Projection GEMM:  out[m,n] = (sum_k X[m,k] * W[n,k] + bias[n]) * scale
// X: [MROWS, EMB] bf16, W: [EMB, EMB] bf16 (row n is contiguous in k).
// Workgroup = 4 waves; each wave owns a 16(M) x 64(N) tile; K-loop step 32.
// mode 0/1: out is [head][t][d] bf16 (head = b*H + h)
// mode 2  : out is [head][d][t] bf16 (V transposed for the P@V B-layout)
// ---------------------------------------------------------------------------
__global__ void __launch_bounds__(128)
proj_kernel(const unsigned short* __restrict__ X,
            const unsigned short* __restrict__ W,
            const float* __restrict__ bias,
            unsigned short* __restrict__ out,
            int mode, float scale) {
  const int wave = threadIdx.x >> 5;
  const int lane = threadIdx.x & 31;
  const int row  = lane & 15;
  const int hi   = lane >> 4;
  const int m0 = blockIdx.x * 64 + wave * 16;
  const int n0 = blockIdx.y * 64;

  v8f acc[4] = {v8f{}, v8f{}, v8f{}, v8f{}};

  const unsigned short* arow = X + (size_t)(m0 + row) * EMB;
  const unsigned short* brow[4];
#pragma unroll
  for (int nt = 0; nt < 4; ++nt)
    brow[nt] = W + (size_t)(n0 + nt * 16 + row) * EMB + hi * 16;

  for (int kc = 0; kc < EMB; kc += 32) {
    v16bf a = load16(arow + kc + hi * 16);
#pragma unroll
    for (int nt = 0; nt < 4; ++nt) {
      v16bf b = load16(brow[nt] + kc);
      acc[nt] = wmma_bf16(a, b, acc[nt]);
    }
  }

#pragma unroll
  for (int nt = 0; nt < 4; ++nt) {
    const int n = n0 + nt * 16 + row;
    const float bv = bias[n];
    const int h = n >> 6, d = n & 63;
#pragma unroll
    for (int r = 0; r < 8; ++r) {
      const int m = m0 + r + hi * 8;
      const float val = (acc[nt][r] + bv) * scale;
      const int t = m >> 2, b = m & 3;          // m = t*B + b (B = 4)
      const int head = b * HEADS + h;
      size_t idx;
      if (mode == 2) idx = ((size_t)(head * HDIM + d)) * T_LEN + t;
      else           idx = ((size_t)head * T_LEN + t) * HDIM + d;
      out[idx] = f2bf(val);
    }
  }
}

// ---------------------------------------------------------------------------
// Flash attention, causal. One wave = 16 query rows of one head.
// K-chunk of 32 keys per iteration:
//   S (2x 16x16 tiles) via 4 WMMAs (D=64 -> two k=32 steps each)
//   online softmax in f32 (shfl-xor row reductions within 16-lane groups)
//   P restaged C-layout -> A-layout through per-wave LDS (in-order DS)
//   O += P @ V via 4 WMMAs (V stored transposed -> contiguous B frags)
// ---------------------------------------------------------------------------
__global__ void __launch_bounds__(128)
flash_attn_kernel(const unsigned short* __restrict__ qh,   // [64][T][64]
                  const unsigned short* __restrict__ kh,   // [64][T][64]
                  const unsigned short* __restrict__ vt,   // [64][64][T]
                  unsigned short* __restrict__ attnout) {  // [MROWS, EMB]
  __shared__ __align__(64) unsigned short ldsP[4][16 * 32];

  const int wave = threadIdx.x >> 5;
  const int lane = threadIdx.x & 31;
  const int row  = lane & 15;
  const int hi   = lane >> 4;
  const int head  = blockIdx.y;
  const int qbase = blockIdx.x * 64 + wave * 16;

  const unsigned short* qrow = qh + ((size_t)head * T_LEN + qbase + row) * HDIM;
  const v16bf aq0 = load16(qrow + hi * 16);
  const v16bf aq1 = load16(qrow + 32 + hi * 16);

  v8f o[4] = {v8f{}, v8f{}, v8f{}, v8f{}};
  float mrun[8], lrun[8];
#pragma unroll
  for (int r = 0; r < 8; ++r) { mrun[r] = -1e30f; lrun[r] = 0.f; }

  unsigned short* myP = &ldsP[wave][0];
  const unsigned short* vbase =
      vt + ((size_t)head * HDIM + row) * T_LEN + hi * 16;

  for (int jc = 0; jc < qbase + 16; jc += 32) {   // causal: only chunks <= diag
    const unsigned short* k0 =
        kh + ((size_t)head * T_LEN + jc + row) * HDIM + hi * 16;
    const unsigned short* k1 = k0 + 16 * HDIM;
    const v16bf bk00 = load16(k0);
    const v16bf bk01 = load16(k0 + 32);
    const v16bf bk10 = load16(k1);
    const v16bf bk11 = load16(k1 + 32);

    v8f s0{}; v8f s1{};
    s0 = wmma_bf16(aq0, bk00, s0);
    s0 = wmma_bf16(aq1, bk01, s0);
    s1 = wmma_bf16(aq0, bk10, s1);
    s1 = wmma_bf16(aq1, bk11, s1);

    const int c0 = jc + row;
    const int c1 = c0 + 16;
#pragma unroll
    for (int r = 0; r < 8; ++r) {
      const int rg = qbase + r + hi * 8;
      float e0 = (c0 <= rg) ? s0[r] : -1e30f;     // causal additive mask
      float e1 = (c1 <= rg) ? s1[r] : -1e30f;
      float mx = fmaxf(e0, e1);
      mx = fmaxf(mx, __shfl_xor(mx, 1, 32));
      mx = fmaxf(mx, __shfl_xor(mx, 2, 32));
      mx = fmaxf(mx, __shfl_xor(mx, 4, 32));
      mx = fmaxf(mx, __shfl_xor(mx, 8, 32));
      const float mnew = fmaxf(mrun[r], mx);
      const float corr = __expf(mrun[r] - mnew);
      const float p0 = __expf(e0 - mnew);
      const float p1 = __expf(e1 - mnew);
      float sm = p0 + p1;
      sm += __shfl_xor(sm, 1, 32);
      sm += __shfl_xor(sm, 2, 32);
      sm += __shfl_xor(sm, 4, 32);
      sm += __shfl_xor(sm, 8, 32);
      lrun[r] = lrun[r] * corr + sm;
      mrun[r] = mnew;
#pragma unroll
      for (int nt = 0; nt < 4; ++nt) o[nt][r] *= corr;
      myP[(r + hi * 8) * 32 + row]      = f2bf(p0);
      myP[(r + hi * 8) * 32 + row + 16] = f2bf(p1);
    }
    __asm__ volatile("" ::: "memory");   // keep DS store->load program order
    const v16bf ap = load16(myP + row * 32 + hi * 16);
#pragma unroll
    for (int nt = 0; nt < 4; ++nt) {
      v16bf bvv = load16(vbase + (size_t)nt * 16 * T_LEN + jc);
      o[nt] = wmma_bf16(ap, bvv, o[nt]);
    }
    __asm__ volatile("" ::: "memory");
  }

  const int b = head >> 4, h = head & 15;
#pragma unroll
  for (int nt = 0; nt < 4; ++nt) {
    const int e = h * HDIM + nt * 16 + row;
#pragma unroll
    for (int r = 0; r < 8; ++r) {
      const int t = qbase + r + hi * 8;
      const float val = o[nt][r] / lrun[r];
      attnout[((size_t)t * BATCH + b) * EMB + e] = f2bf(val);
    }
  }
}

// ---------------------------------------------------------------------------
// Output projection: f32 result straight to d_out ([T,B,E] == [MROWS, EMB])
// ---------------------------------------------------------------------------
__global__ void __launch_bounds__(128)
outproj_kernel(const unsigned short* __restrict__ X,   // attn [MROWS, EMB] bf16
               const unsigned short* __restrict__ W,   // wo bf16
               const float* __restrict__ bias,
               float* __restrict__ out) {
  const int wave = threadIdx.x >> 5;
  const int lane = threadIdx.x & 31;
  const int row  = lane & 15;
  const int hi   = lane >> 4;
  const int m0 = blockIdx.x * 64 + wave * 16;
  const int n0 = blockIdx.y * 64;

  v8f acc[4] = {v8f{}, v8f{}, v8f{}, v8f{}};

  const unsigned short* arow = X + (size_t)(m0 + row) * EMB;
  const unsigned short* brow[4];
#pragma unroll
  for (int nt = 0; nt < 4; ++nt)
    brow[nt] = W + (size_t)(n0 + nt * 16 + row) * EMB + hi * 16;

  for (int kc = 0; kc < EMB; kc += 32) {
    v16bf a = load16(arow + kc + hi * 16);
#pragma unroll
    for (int nt = 0; nt < 4; ++nt) {
      v16bf b = load16(brow[nt] + kc);
      acc[nt] = wmma_bf16(a, b, acc[nt]);
    }
  }

#pragma unroll
  for (int nt = 0; nt < 4; ++nt) {
    const int n = n0 + nt * 16 + row;
    const float bv = bias[n];
#pragma unroll
    for (int r = 0; r < 8; ++r) {
      const int m = m0 + r + hi * 8;
      out[(size_t)m * EMB + n] = acc[nt][r] + bv;
    }
  }
}

// ---------------------------------------------------------------------------
extern "C" void kernel_launch(void* const* d_in, const int* in_sizes, int n_in,
                              void* d_out, int out_size, void* d_ws, size_t ws_size,
                              hipStream_t stream) {
  (void)in_sizes; (void)n_in; (void)out_size; (void)ws_size;

  const float* query = (const float*)d_in[0];
  const float* key_  = (const float*)d_in[1];
  const float* value = (const float*)d_in[2];
  // d_in[3] = attn_mask: causal -1e9; implemented analytically in the kernel.
  const float* wq = (const float*)d_in[4];
  const float* bq = (const float*)d_in[5];
  const float* wk = (const float*)d_in[6];
  const float* bk = (const float*)d_in[7];
  const float* wv = (const float*)d_in[8];
  const float* bv = (const float*)d_in[9];
  const float* wo = (const float*)d_in[10];
  const float* bo = (const float*)d_in[11];
  float* out = (float*)d_out;

  const size_t ME = (size_t)MROWS * EMB;            // 8M elems
  const size_t EE = (size_t)EMB * EMB;              // 1M elems
  const size_t HTD = (size_t)NHEAD * T_LEN * HDIM;  // 8M elems

  char* ws = (char*)d_ws;
  size_t off = 0;
  unsigned short* xq  = (unsigned short*)(ws + off); off += ME * 2;
  unsigned short* xk  = (unsigned short*)(ws + off); off += ME * 2;
  unsigned short* xv  = (unsigned short*)(ws + off); off += ME * 2;
  unsigned short* wqb = (unsigned short*)(ws + off); off += EE * 2;
  unsigned short* wkb = (unsigned short*)(ws + off); off += EE * 2;
  unsigned short* wvb = (unsigned short*)(ws + off); off += EE * 2;
  unsigned short* wob = (unsigned short*)(ws + off); off += EE * 2;
  unsigned short* qhb = (unsigned short*)(ws + off); off += HTD * 2;
  unsigned short* khb = (unsigned short*)(ws + off); off += HTD * 2;
  unsigned short* vtb = (unsigned short*)(ws + off); off += HTD * 2;
  unsigned short* att = (unsigned short*)(ws + off); off += ME * 2;

  const int nME = (int)ME, nEE = (int)EE;
  cvt_bf16_kernel<<<dim3((nME + 255) / 256), dim3(256), 0, stream>>>(query, xq, nME);
  cvt_bf16_kernel<<<dim3((nME + 255) / 256), dim3(256), 0, stream>>>(key_,  xk, nME);
  cvt_bf16_kernel<<<dim3((nME + 255) / 256), dim3(256), 0, stream>>>(value, xv, nME);
  cvt_bf16_kernel<<<dim3((nEE + 255) / 256), dim3(256), 0, stream>>>(wq, wqb, nEE);
  cvt_bf16_kernel<<<dim3((nEE + 255) / 256), dim3(256), 0, stream>>>(wk, wkb, nEE);
  cvt_bf16_kernel<<<dim3((nEE + 255) / 256), dim3(256), 0, stream>>>(wv, wvb, nEE);
  cvt_bf16_kernel<<<dim3((nEE + 255) / 256), dim3(256), 0, stream>>>(wo, wob, nEE);

  dim3 gp(MROWS / 64, EMB / 64);
  proj_kernel<<<gp, dim3(128), 0, stream>>>(xq, wqb, bq, qhb, 0, 0.125f); // D^-1/2
  proj_kernel<<<gp, dim3(128), 0, stream>>>(xk, wkb, bk, khb, 1, 1.0f);
  proj_kernel<<<gp, dim3(128), 0, stream>>>(xv, wvb, bv, vtb, 2, 1.0f);

  flash_attn_kernel<<<dim3(T_LEN / 64, NHEAD), dim3(128), 0, stream>>>(
      qhb, khb, vtb, att);

  outproj_kernel<<<gp, dim3(128), 0, stream>>>(att, wob, bo, out);
}
